// Model_26439818674684
// MI455X (gfx1250) — compile-verified, matrix operand order checked by reading, not run on hardware
//
#include <hip/hip_runtime.h>

// ---------------------------------------------------------------------------
// Model sizes
// ---------------------------------------------------------------------------
#define DAYS   30
#define NTOP   20
#define TLEN   128
#define EDIM   300
#define HDIM   256
#define BATCH  600          // DAYS * NTOP
#define BPAD   608          // 19 * 32
#define BPAD2  640          // covers topic pad-row strided reads (32*20)

typedef __attribute__((ext_vector_type(16))) __bf16       v16bf;
typedef __attribute__((ext_vector_type(8)))  float        v8f;
typedef __attribute__((ext_vector_type(4)))  unsigned int u32x4;

union Frag32 { v16bf v; u32x4 u[2]; };

__device__ __forceinline__ unsigned short f2bfu(float f) {
  unsigned int x = __float_as_uint(f);
  x += 0x7FFFu + ((x >> 16) & 1u);          // round-to-nearest-even
  return (unsigned short)(x >> 16);
}
__device__ __forceinline__ float fast_exp(float x) {
  return __builtin_amdgcn_exp2f(x * 1.4426950408889634f);
}
__device__ __forceinline__ float fast_rcp(float x) { return __builtin_amdgcn_rcpf(x); }
__device__ __forceinline__ float sigmf(float x) { return fast_rcp(1.0f + fast_exp(-x)); }
__device__ __forceinline__ float tanh_f(float x) {
  return 1.0f - 2.0f * fast_rcp(1.0f + fast_exp(2.0f * x));
}

// ---------------------------------------------------------------------------
// Convert a [N=4H, K] fp32 weight matrix into bf16 WMMA B-fragment layout:
//   dst[((nt*Kc + kt)*32 + lane)*16 + j] = W[n, k]
//   n = nt*16 + (lane&15)
//   k = kt*32 + (j&8 ? 16 : 0) + (lane>>4)*8 + (j&7)      (zero-pad k >= K)
// ---------------------------------------------------------------------------
__global__ void convert_wfrag(const float* __restrict__ W,
                              unsigned short* __restrict__ dst,
                              int K, int Kc, int total) {
  int idx = blockIdx.x * blockDim.x + threadIdx.x;
  if (idx >= total) return;
  int j    = idx & 15;
  int lane = (idx >> 4) & 31;
  int q    = idx >> 9;
  int kt   = q % Kc;
  int nt   = q / Kc;
  int n = nt * 16 + (lane & 15);
  int k = kt * 32 + ((j & 8) ? 16 : 0) + ((lane >> 4) * 8) + (j & 7);
  float v = (k < K) ? W[(long long)n * K + k] : 0.0f;
  dst[idx] = f2bfu(v);
}

// ---------------------------------------------------------------------------
// One LSTM timestep for a 32-row batch tile, H = 256 (4H = 1024 gates).
// gates = x_t @ Wx^T + h @ Wh^T + b ; then cell update in registers.
// Wave w owns N-tiles { g*256 + p*128 + w*16 } so it holds all 4 gates for
// its hidden units. x comes either from fp32 (LDS cvt-stage, double-buffered)
// or from a row-major bf16 buffer read directly as A fragments. The recurrent
// term always reads the bf16 hidden state (hbf) directly from global.
// ---------------------------------------------------------------------------
__global__ __launch_bounds__(256) void lstm_step_wmma(
    const float* __restrict__ xtF,           // fp32 x (or null)
    const unsigned short* __restrict__ xtB,  // bf16 x rows (or null)
    long long xRowStride, int Kx, int KcX,
    const unsigned short* __restrict__ WxF,
    const unsigned short* __restrict__ WhF,
    const float* __restrict__ bias,
    float* __restrict__ h, unsigned short* __restrict__ hbf,
    float* __restrict__ c,
    float* __restrict__ yout, unsigned short* __restrict__ youtbf,
    int youtRowStride, int nRowsValid)
{
  __shared__ __align__(16) unsigned short ldsA[2][32 * 40];  // double buffer
  const int tid  = threadIdx.x;
  const int lane = tid & 31;
  const int wv   = tid >> 5;
  const int hlf  = lane >> 4;
  const int l15  = lane & 15;
  const int rowBase = blockIdx.x * 32;

  v8f acc[2][8];
#pragma unroll
  for (int mt = 0; mt < 2; ++mt)
#pragma unroll
    for (int s = 0; s < 8; ++s)
#pragma unroll
      for (int e = 0; e < 8; ++e) acc[mt][s][e] = 0.0f;

  Frag32 af[2];

  if (xtB == nullptr) {
    // ---- phase 0a: fp32 x, LDS convert-stage, double buffered ----
    const int r    = tid >> 3;
    const int kq   = (tid & 7) << 2;
    const int grow = rowBase + r;
#pragma unroll
    for (int i2 = 0; i2 < 4; ++i2) {   // stage chunk 0
      const int k = kq + i2;
      float v = 0.0f;
      if (grow < nRowsValid && k < Kx) v = xtF[(long long)grow * xRowStride + k];
      ldsA[0][r * 40 + kq + i2] = f2bfu(v);
    }
    for (int kt = 0; kt < KcX; ++kt) {
      __syncthreads();
      if (kt + 1 < KcX) {              // prefetch next chunk into other buffer
        const int kb = (kt + 1) * 32 + kq;
#pragma unroll
        for (int i2 = 0; i2 < 4; ++i2) {
          const int k = kb + i2;
          float v = 0.0f;
          if (grow < nRowsValid && k < Kx) v = xtF[(long long)grow * xRowStride + k];
          ldsA[(kt + 1) & 1][r * 40 + kq + i2] = f2bfu(v);
        }
      }
      const unsigned short* bufp = ldsA[kt & 1];
#pragma unroll
      for (int mt = 0; mt < 2; ++mt) {
        const unsigned short* p0 = &bufp[(mt * 16 + l15) * 40 + hlf * 8];
        af[mt].u[0] = *(const u32x4*)(p0);
        af[mt].u[1] = *(const u32x4*)(p0 + 16);
      }
#pragma unroll
      for (int s = 0; s < 8; ++s) {
        const int ntile = (s & 1) * 8 + (s >> 1) * 16 + wv;
        const unsigned short* bp = WxF + ((ntile * KcX + kt) * 32 + lane) * 16;
        Frag32 bfr;
        bfr.u[0] = *(const u32x4*)(bp);
        bfr.u[1] = *(const u32x4*)(bp + 8);
#pragma unroll
        for (int mt = 0; mt < 2; ++mt)
          acc[mt][s] = __builtin_amdgcn_wmma_f32_16x16x32_bf16(
              false, af[mt].v, false, bfr.v, (short)0, acc[mt][s], false, false);
      }
    }
  } else {
    // ---- phase 0b: bf16 x, direct global A fragments ----
    for (int kt = 0; kt < KcX; ++kt) {
#pragma unroll
      for (int mt = 0; mt < 2; ++mt) {
        const unsigned short* p0 =
            xtB + (long long)(rowBase + mt * 16 + l15) * xRowStride + kt * 32 + hlf * 8;
        af[mt].u[0] = *(const u32x4*)(p0);
        af[mt].u[1] = *(const u32x4*)(p0 + 16);
      }
#pragma unroll
      for (int s = 0; s < 8; ++s) {
        const int ntile = (s & 1) * 8 + (s >> 1) * 16 + wv;
        const unsigned short* bp = WxF + ((ntile * KcX + kt) * 32 + lane) * 16;
        Frag32 bfr;
        bfr.u[0] = *(const u32x4*)(bp);
        bfr.u[1] = *(const u32x4*)(bp + 8);
#pragma unroll
        for (int mt = 0; mt < 2; ++mt)
          acc[mt][s] = __builtin_amdgcn_wmma_f32_16x16x32_bf16(
              false, af[mt].v, false, bfr.v, (short)0, acc[mt][s], false, false);
      }
    }
  }

  // ---- phase 1: recurrent term, bf16 hidden state direct from global ----
  for (int kt = 0; kt < 8; ++kt) {
#pragma unroll
    for (int mt = 0; mt < 2; ++mt) {
      const unsigned short* p0 =
          hbf + (long long)(rowBase + mt * 16 + l15) * 256 + kt * 32 + hlf * 8;
      af[mt].u[0] = *(const u32x4*)(p0);
      af[mt].u[1] = *(const u32x4*)(p0 + 16);
    }
#pragma unroll
    for (int s = 0; s < 8; ++s) {
      const int ntile = (s & 1) * 8 + (s >> 1) * 16 + wv;
      const unsigned short* bp = WhF + ((ntile * 8 + kt) * 32 + lane) * 16;
      Frag32 bfr;
      bfr.u[0] = *(const u32x4*)(bp);
      bfr.u[1] = *(const u32x4*)(bp + 8);
#pragma unroll
      for (int mt = 0; mt < 2; ++mt)
        acc[mt][s] = __builtin_amdgcn_wmma_f32_16x16x32_bf16(
            false, af[mt].v, false, bfr.v, (short)0, acc[mt][s], false, false);
    }
  }

  // all waves' hbf reads are complete (WMMA consumed them); order them
  // against the hbf writes below.
  __syncthreads();

  // ---- bias + LSTM cell update, in registers ----
  float bval[8];
#pragma unroll
  for (int s = 0; s < 8; ++s) {
    const int n = (s & 1) * 128 + (s >> 1) * 256 + wv * 16 + l15;
    bval[s] = bias[n];
  }
#pragma unroll
  for (int p = 0; p < 2; ++p) {
    const int u = p * 128 + wv * 16 + l15;
#pragma unroll
    for (int mt = 0; mt < 2; ++mt) {
#pragma unroll
      for (int e = 0; e < 8; ++e) {
        const int row = rowBase + mt * 16 + hlf * 8 + e;
        const float gi = sigmf(acc[mt][0 + p][e] + bval[0 + p]);
        const float gf = sigmf(acc[mt][2 + p][e] + bval[2 + p]);
        const float gg = tanh_f(acc[mt][4 + p][e] + bval[4 + p]);
        const float go = sigmf(acc[mt][6 + p][e] + bval[6 + p]);
        const long long off = (long long)row * 256 + u;
        const float cn = gf * c[off] + gi * gg;
        const float hn = go * tanh_f(cn);
        c[off]   = cn;
        h[off]   = hn;
        hbf[off] = f2bfu(hn);
        if (yout)   yout[(long long)row * youtRowStride + u] = hn;
        if (youtbf) youtbf[(long long)row * youtRowStride + u] = f2bfu(hn);
      }
    }
  }
}

// ---------------------------------------------------------------------------
// Day-level LSTM step (batch=1, DH=64, 4DH=256): fp32 GEMV + update.
// ---------------------------------------------------------------------------
__global__ __launch_bounds__(256) void day_lstm_step(
    const float* __restrict__ xt, int K,
    const float* __restrict__ Wih, const float* __restrict__ Whh,
    const float* __restrict__ bias,
    float* __restrict__ h, float* __restrict__ c, float* __restrict__ yout)
{
  __shared__ float g[256];
  __shared__ float hs[64];
  const int tid = threadIdx.x;
  if (tid < 64) hs[tid] = h[tid];
  __syncthreads();
  float a = bias[tid];
  const float* wr = Wih + tid * K;
  for (int k = 0; k < K; ++k) a += wr[k] * xt[k];
  const float* ur = Whh + tid * 64;
  for (int k = 0; k < 64; ++k) a += ur[k] * hs[k];
  g[tid] = a;
  __syncthreads();
  if (tid < 64) {
    const float gi = sigmf(g[tid]);
    const float gf = sigmf(g[64 + tid]);
    const float gg = tanh_f(g[128 + tid]);
    const float go = sigmf(g[192 + tid]);
    const float cn = gf * c[tid] + gi * gg;
    const float hn = go * tanh_f(cn);
    c[tid] = cn; h[tid] = hn; yout[tid] = hn;
  }
}

// ---------------------------------------------------------------------------
// Per-day topic attention with top-p (0.8) truncation.
// score_j = y_top1[d,j,:] . (w1_W^T h_top0[d])   (w1_b term softmax-invariant)
// ---------------------------------------------------------------------------
__global__ __launch_bounds__(256) void topic_attn(
    const float* __restrict__ y_top1,   // [32,20,256] (rows >=30 padded)
    const float* __restrict__ h_top0,   // [32,256]
    const float* __restrict__ w1_W,     // [256,256]
    float* __restrict__ day_hidden)     // [32,256]
{
  const int d = blockIdx.x;
  __shared__ float m[256];
  __shared__ float fin[256];
  __shared__ float sc[20];
  __shared__ float coef[20];
  const int tid = threadIdx.x;
  fin[tid] = h_top0[d * 256 + tid];
  __syncthreads();
  float a = 0.0f;
  for (int u = 0; u < 256; ++u) a += w1_W[u * 256 + tid] * fin[u];
  m[tid] = a;
  __syncthreads();
  if (tid < 20) {
    const float* row = y_top1 + ((long long)d * 20 + tid) * 256;
    float s = 0.0f;
    for (int k = 0; k < 256; ++k) s += row[k] * m[k];
    sc[tid] = s;
  }
  __syncthreads();
  if (tid == 0) {
    float mx = sc[0];
    for (int j = 1; j < 20; ++j) mx = fmaxf(mx, sc[j]);
    float e[20]; float sum = 0.0f;
    for (int j = 0; j < 20; ++j) { e[j] = fast_exp(sc[j] - mx); sum += e[j]; }
    const float inv = 1.0f / sum;
    float p[20]; int idx[20];
    for (int j = 0; j < 20; ++j) { p[j] = e[j] * inv; idx[j] = j; }
    for (int a1 = 1; a1 < 20; ++a1) {          // insertion sort, descending
      float pv = p[a1]; int iv = idx[a1]; int b1 = a1 - 1;
      while (b1 >= 0 && p[b1] < pv) { p[b1 + 1] = p[b1]; idx[b1 + 1] = idx[b1]; --b1; }
      p[b1 + 1] = pv; idx[b1 + 1] = iv;
    }
    for (int j = 0; j < 20; ++j) coef[j] = 0.0f;
    float cum = 0.0f;
    for (int j = 0; j < 20; ++j) {             // keep while exclusive-cumsum <= 0.8
      if (cum <= 0.8f) coef[idx[j]] = p[j];
      cum += p[j];
    }
  }
  __syncthreads();
  float o = 0.0f;
  for (int j = 0; j < 20; ++j)
    o += coef[j] * y_top1[((long long)d * 20 + j) * 256 + tid];
  day_hidden[d * 256 + tid] = o;
}

// ---------------------------------------------------------------------------
// Day attention + linear head -> out[4]
// ---------------------------------------------------------------------------
__global__ __launch_bounds__(256) void day_head(
    const float* __restrict__ y_day1,  // [30,64]
    const float* __restrict__ h_day0,  // [64]
    const float* __restrict__ w2_W,    // [64,64]
    const float* __restrict__ lin1_W, const float* __restrict__ lin1_b,
    const float* __restrict__ lin2_W, const float* __restrict__ lin2_b,
    const float* __restrict__ head_W, const float* __restrict__ head_b,
    const float* __restrict__ prev,    // [4,4]
    float* __restrict__ out)           // [4]
{
  __shared__ float m[64], at[30], ctx[64], h1[48], h2[16];
  const int tid = threadIdx.x;
  if (tid < 64) {
    float a = 0.0f;
    for (int u = 0; u < 64; ++u) a += w2_W[u * 64 + tid] * h_day0[u];
    m[tid] = a;
  }
  __syncthreads();
  if (tid < 30) {
    float s = 0.0f;
    const float* r = y_day1 + tid * 64;
    for (int k = 0; k < 64; ++k) s += r[k] * m[k];
    at[tid] = s;
  }
  __syncthreads();
  if (tid == 0) {
    float mx = at[0];
    for (int j = 1; j < 30; ++j) mx = fmaxf(mx, at[j]);
    float sum = 0.0f;
    for (int j = 0; j < 30; ++j) { at[j] = fast_exp(at[j] - mx); sum += at[j]; }
    const float inv = 1.0f / sum;
    for (int j = 0; j < 30; ++j) at[j] *= inv;
  }
  __syncthreads();
  if (tid < 64) {
    float a = 0.0f;
    for (int t = 0; t < 30; ++t) a += at[t] * y_day1[t * 64 + tid];
    ctx[tid] = a;
  }
  __syncthreads();
  if (tid < 48) {
    float a = lin1_b[tid];
    for (int k = 0; k < 64; ++k) a += lin1_W[tid * 64 + k] * ctx[k];
    h1[tid] = a;
  }
  __syncthreads();
  if (tid < 16) {
    float a = lin2_b[tid];
    for (int k = 0; k < 48; ++k) a += lin2_W[tid * 48 + k] * h1[k];
    h2[tid] = a;
  }
  __syncthreads();
  if (tid < 4) {
    float a = head_b[tid];
    for (int cc = 0; cc < 16; ++cc) a += h2[cc] * head_W[tid * 20 + cc];
    for (int cc = 0; cc < 4; ++cc)  a += prev[tid * 4 + cc] * head_W[tid * 20 + 16 + cc];
    out[tid] = a;
  }
}

// ---------------------------------------------------------------------------
extern "C" void kernel_launch(void* const* d_in, const int* in_sizes, int n_in,
                              void* d_out, int out_size, void* d_ws, size_t ws_size,
                              hipStream_t stream) {
  (void)in_sizes; (void)n_in; (void)out_size; (void)ws_size;

  const float* X        = (const float*)d_in[0];
  const float* prev     = (const float*)d_in[1];
  const float* txt_Wih0 = (const float*)d_in[2];
  const float* txt_Whh0 = (const float*)d_in[3];
  const float* txt_b0   = (const float*)d_in[4];
  // d_in[5..7] = text layer-1 weights: dead code in the reference
  const float* top_Wih0 = (const float*)d_in[8];
  const float* top_Whh0 = (const float*)d_in[9];
  const float* top_b0   = (const float*)d_in[10];
  const float* top_Wih1 = (const float*)d_in[11];
  const float* top_Whh1 = (const float*)d_in[12];
  const float* top_b1   = (const float*)d_in[13];
  const float* day_Wih0 = (const float*)d_in[14];
  const float* day_Whh0 = (const float*)d_in[15];
  const float* day_b0   = (const float*)d_in[16];
  const float* day_Wih1 = (const float*)d_in[17];
  const float* day_Whh1 = (const float*)d_in[18];
  const float* day_b1   = (const float*)d_in[19];
  const float* w1_W     = (const float*)d_in[20];
  // d_in[21] w1_b, d_in[23] w2_b: softmax-invariant, dropped
  const float* w2_W     = (const float*)d_in[22];
  const float* lin1_W   = (const float*)d_in[24];
  const float* lin1_b   = (const float*)d_in[25];
  const float* lin2_W   = (const float*)d_in[26];
  const float* lin2_b   = (const float*)d_in[27];
  const float* head_W   = (const float*)d_in[28];
  const float* head_b   = (const float*)d_in[29];

  // ---- workspace carve-up (256B aligned regions) ----
  unsigned char* base = (unsigned char*)d_ws;
  size_t off = 0;
  auto alloc = [&](size_t bytes) -> void* {
    void* p = base + off;
    off = (off + bytes + 255) & ~(size_t)255;
    return p;
  };
  unsigned short* wxT  = (unsigned short*)alloc((size_t)64 * 10 * 512 * 2); // txt Wih0, Kc=10
  unsigned short* whT  = (unsigned short*)alloc((size_t)64 * 8  * 512 * 2); // txt Whh0
  unsigned short* wx0  = (unsigned short*)alloc((size_t)64 * 8  * 512 * 2); // top Wih0
  unsigned short* wh0  = (unsigned short*)alloc((size_t)64 * 8  * 512 * 2); // top Whh0
  unsigned short* wx1  = (unsigned short*)alloc((size_t)64 * 8  * 512 * 2); // top Wih1
  unsigned short* wh1  = (unsigned short*)alloc((size_t)64 * 8  * 512 * 2); // top Whh1
  float*          hT   = (float*)alloc((size_t)BPAD * 256 * 4);
  float*          cT   = (float*)alloc((size_t)BPAD * 256 * 4);
  unsigned short* hbfT = (unsigned short*)alloc((size_t)BPAD2 * 256 * 2);   // bf16 text h
  float*          h0t  = (float*)alloc((size_t)32 * 256 * 4);
  float*          c0t  = (float*)alloc((size_t)32 * 256 * 4);
  unsigned short* hbf0 = (unsigned short*)alloc((size_t)32 * 256 * 2);
  float*          h1t  = (float*)alloc((size_t)32 * 256 * 4);
  float*          c1t  = (float*)alloc((size_t)32 * 256 * 4);
  unsigned short* hbf1 = (unsigned short*)alloc((size_t)32 * 256 * 2);
  unsigned short* y0bf = (unsigned short*)alloc((size_t)32 * 20 * 256 * 2); // topic l0 out, bf16
  float*          y1t  = (float*)alloc((size_t)32 * 20 * 256 * 4);          // topic l1 out, fp32
  float*          dayH = (float*)alloc((size_t)32 * 256 * 4);
  float*          dh0  = (float*)alloc(256 * 4);
  float*          dc0  = (float*)alloc(256 * 4);
  float*          dh1  = (float*)alloc(256 * 4);
  float*          dc1  = (float*)alloc(256 * 4);
  float*          yd0  = (float*)alloc((size_t)32 * 64 * 4);
  float*          yd1  = (float*)alloc((size_t)32 * 64 * 4);

  // ---- zero-init recurrent state (every call; graph-capture safe) ----
  hipMemsetAsync(hT,   0, (size_t)BPAD * 256 * 4, stream);
  hipMemsetAsync(cT,   0, (size_t)BPAD * 256 * 4, stream);
  hipMemsetAsync(hbfT, 0, (size_t)BPAD2 * 256 * 2, stream);
  hipMemsetAsync(h0t,  0, (size_t)32 * 256 * 4, stream);
  hipMemsetAsync(c0t,  0, (size_t)32 * 256 * 4, stream);
  hipMemsetAsync(hbf0, 0, (size_t)32 * 256 * 2, stream);
  hipMemsetAsync(h1t,  0, (size_t)32 * 256 * 4, stream);
  hipMemsetAsync(c1t,  0, (size_t)32 * 256 * 4, stream);
  hipMemsetAsync(hbf1, 0, (size_t)32 * 256 * 2, stream);
  hipMemsetAsync(dh0,  0, 256 * 4, stream);
  hipMemsetAsync(dc0,  0, 256 * 4, stream);
  hipMemsetAsync(dh1,  0, 256 * 4, stream);
  hipMemsetAsync(dc1,  0, 256 * 4, stream);

  // ---- weight fragment conversion (fp32 -> bf16 B-fragments) ----
  auto cvt = [&](const float* W, unsigned short* dst, int K, int Kc) {
    int total = 64 * Kc * 512;
    convert_wfrag<<<(total + 255) / 256, 256, 0, stream>>>(W, dst, K, Kc, total);
  };
  cvt(txt_Wih0, wxT, 300, 10);
  cvt(txt_Whh0, whT, 256, 8);
  cvt(top_Wih0, wx0, 256, 8);
  cvt(top_Whh0, wh0, 256, 8);
  cvt(top_Wih1, wx1, 256, 8);
  cvt(top_Whh1, wh1, 256, 8);

  // ---- text LSTM layer 0 (layer 1 is dead code): 128 steps, batch 600 ----
  for (int t = 0; t < TLEN; ++t) {
    lstm_step_wmma<<<BPAD / 32, 256, 0, stream>>>(
        X + (long long)t * EDIM, nullptr, (long long)TLEN * EDIM, EDIM, 10,
        wxT, whT, txt_b0, hT, hbfT, cT, nullptr, nullptr, 0, BATCH);
  }

  // ---- topic LSTM layer 0: x = bf16 text hidden, strided rows ----
  for (int t = 0; t < NTOP; ++t) {
    lstm_step_wmma<<<1, 256, 0, stream>>>(
        nullptr, hbfT + (long long)t * 256, (long long)NTOP * 256, 256, 8,
        wx0, wh0, top_b0, h0t, hbf0, c0t,
        nullptr, y0bf + (long long)t * 256, NTOP * 256, DAYS);
  }
  // ---- topic LSTM layer 1: x = bf16 layer-0 outputs ----
  for (int t = 0; t < NTOP; ++t) {
    lstm_step_wmma<<<1, 256, 0, stream>>>(
        nullptr, y0bf + (long long)t * 256, (long long)NTOP * 256, 256, 8,
        wx1, wh1, top_b1, h1t, hbf1, c1t,
        y1t + (long long)t * 256, nullptr, NTOP * 256, DAYS);
  }

  // ---- per-day topic attention with top-p truncation ----
  topic_attn<<<DAYS, 256, 0, stream>>>(y1t, h0t, w1_W, dayH);

  // ---- day LSTM (batch 1, DH=64), layer 0 then layer 1 ----
  for (int t = 0; t < DAYS; ++t)
    day_lstm_step<<<1, 256, 0, stream>>>(dayH + t * 256, 256, day_Wih0, day_Whh0,
                                         day_b0, dh0, dc0, yd0 + t * 64);
  for (int t = 0; t < DAYS; ++t)
    day_lstm_step<<<1, 256, 0, stream>>>(yd0 + t * 64, 64, day_Wih1, day_Whh1,
                                         day_b1, dh1, dc1, yd1 + t * 64);

  // ---- day attention + head -> out[4] ----
  day_head<<<1, 256, 0, stream>>>(yd1, dh0, w2_W, lin1_W, lin1_b, lin2_W, lin2_b,
                                  head_W, head_b, prev, (float*)d_out);
}
